// STPT_83940840833116
// MI455X (gfx1250) — matrix operand order repaired and checked by reading, not compiled, not run on hardware
//
#include <hip/hip_runtime.h>
#include <hip/hip_fp16.h>

// ---------------------------------------------------------------------------
// Swin-style 3D window attention, fused for MI455X (gfx1250, wave32, WMMA).
//   q:(256,256,256) kv:(256,256,256) mask:(128,256,256) f32
//   out = proj( softmax( scale*QK^T + bias + mask ) @ V )
// Fused per (b,h,qblock) workgroup: Q/K/V projections + attention + softmax
// + attn@V all in LDS; only the 32MB f16 head-context goes to workspace.
// All matmuls via v_wmma_f32_16x16x32_f16 (hd=32 -> QK^T tile = 1 WMMA).
// ---------------------------------------------------------------------------

#define DIM   256
#define HEADS 8
#define HD    32
#define N1    256
#define N2    256
#define NW    128
#define BATCH 256
#define SCALE 0.17677669529663687f   /* 1/sqrt(32) */

typedef __attribute__((ext_vector_type(16))) _Float16 v16h;
typedef __attribute__((ext_vector_type(8)))  _Float16 v8h;
typedef __attribute__((ext_vector_type(8)))  float    v8f;

// D = A(16x32 f16) x B(32x16 f16) + C (f32)
__device__ __forceinline__ v8f wmma16(v16h a, v16h b, v8f c) {
    return __builtin_amdgcn_wmma_f32_16x16x32_f16(
        /*neg_a=*/false, a, /*neg_b=*/false, b,
        /*c_mod=*/(short)0, c, /*reuse_a=*/false, /*reuse_b=*/false);
}

// A fragment (16x32, M rows in lanes) from row-major f32 source, cvt to f16.
// lane r=lane&15 holds row r; half hs=lane>>4: halves[0..15]=K hs*16 .. hs*16+15
__device__ __forceinline__ v16h frag_a_f32(const float* base, int ld, int lane) {
    int r = lane & 15, hs = lane >> 4;
    const float* p = base + (size_t)r * ld + hs * 16;
    v16h out;
#pragma unroll
    for (int i = 0; i < 8; ++i) {
        out[i]     = (_Float16)p[i];
        out[8 + i] = (_Float16)p[8 + i];
    }
    return out;
}

// A fragment from row-major f16 source (LDS or global).
__device__ __forceinline__ v16h frag_a_f16(const _Float16* base, int ld, int lane) {
    int r = lane & 15, hs = lane >> 4;
    const _Float16* p = base + (size_t)r * ld + hs * 16;
    v8h lo = *(const v8h*)p;
    v8h hi = *(const v8h*)(p + 8);
    v16h out;
#pragma unroll
    for (int i = 0; i < 8; ++i) { out[i] = lo[i]; out[8 + i] = hi[i]; }
    return out;
}

// B fragment (32x16, N cols in lanes) where source rows are the N dimension
// and columns are K (i.e. operand used as B^T-style: out = A x Rows^T).
// lane n holds source row n; halves[0..15] = K hs*16 .. hs*16+15.
__device__ __forceinline__ v16h frag_b_f32(const float* base, int ld, int lane) {
    int n = lane & 15, hs = lane >> 4;
    const float* p = base + (size_t)n * ld + hs * 16;
    v16h out;
#pragma unroll
    for (int i = 0; i < 16; ++i) out[i] = (_Float16)p[i];
    return out;
}

__device__ __forceinline__ v16h frag_b_f16(const _Float16* base, int ld, int lane) {
    int n = lane & 15, hs = lane >> 4;
    return *(const v16h*)(base + (size_t)n * ld + hs * 16);
}

// ---------------------------------------------------------------------------
// Prep: bias_f[h][q][k] = bias_table[rel_index[q][k]][h]   (8 x 256 x 256 f32)
// ---------------------------------------------------------------------------
__global__ void bias_gather_kernel(const float* __restrict__ table,
                                   const int* __restrict__ relidx,
                                   float* __restrict__ bias_f) {
    int t = blockIdx.x * 256 + threadIdx.x;       // t in [0, 65536)
    int idx = relidx[t];
    const float* src = table + (size_t)idx * HEADS;
#pragma unroll
    for (int h = 0; h < HEADS; ++h)
        bias_f[(size_t)h * (N1 * N2) + t] = src[h];
}

// ---------------------------------------------------------------------------
// Fused attention. grid = (2 qblocks, HEADS, BATCH), block = 256 (8 wave32).
// Dynamic LDS carve (halves):
//   s_kh [256][32]   K head-proj          8192 h  (16 KB)
//   s_vT [32][256]   V head-proj, transposed 8192 h (16 KB)
//   s_qh [128][32]   Q head-proj (this qblock) 4096 h (8 KB)
//   s_p  [128][256]  normalized softmax probs  32768 h (64 KB)
// ---------------------------------------------------------------------------
__global__ void attn_kernel(const float* __restrict__ q,
                            const float* __restrict__ kv,
                            const float* __restrict__ mask,
                            const float* __restrict__ Wq,
                            const float* __restrict__ bq,
                            const float* __restrict__ Wkv,
                            const float* __restrict__ bkv,
                            const float* __restrict__ bias_f,
                            _Float16* __restrict__ xh) {
    extern __shared__ _Float16 smem[];
    _Float16* s_kh = smem;               // 8192
    _Float16* s_vT = smem + 8192;        // 8192
    _Float16* s_qh = smem + 16384;       // 4096
    _Float16* s_p  = smem + 20480;       // 32768

    const int qb = blockIdx.x;           // 0..1 (128 q rows each)
    const int h  = blockIdx.y;
    const int b  = blockIdx.z;
    const int wv = threadIdx.x >> 5;     // wave id 0..7
    const int lane  = threadIdx.x & 31;
    const int nlane = lane & 15;
    const int hs    = lane >> 4;
    const int wi    = b & (NW - 1);      // mask window index (B/nW = 2)

    // ---------------- Phase 0: K,V head projection (all 256 kv rows) -------
    // wave wv owns kv rows [wv*32, wv*32+32); output tiles 2(rt) x 2(ct).
    {
        v8f accK[2][2] = {}; v8f accV[2][2] = {};
        const float* kvb = kv + (size_t)b * N2 * DIM;
#pragma unroll 2
        for (int c0 = 0; c0 < DIM; c0 += 32) {
            v16h aK[2], bK[2], bV[2];
#pragma unroll
            for (int rt = 0; rt < 2; ++rt)
                aK[rt] = frag_a_f32(kvb + (size_t)(wv * 32 + rt * 16) * DIM + c0, DIM, lane);
#pragma unroll
            for (int ct = 0; ct < 2; ++ct) {
                bK[ct] = frag_b_f32(Wkv + (size_t)(h * HD + ct * 16) * DIM + c0, DIM, lane);
                bV[ct] = frag_b_f32(Wkv + (size_t)(DIM + h * HD + ct * 16) * DIM + c0, DIM, lane);
            }
#pragma unroll
            for (int rt = 0; rt < 2; ++rt)
#pragma unroll
                for (int ct = 0; ct < 2; ++ct) {
                    accK[rt][ct] = wmma16(aK[rt], bK[ct], accK[rt][ct]);
                    accV[rt][ct] = wmma16(aK[rt], bV[ct], accV[rt][ct]);
                }
        }
#pragma unroll
        for (int rt = 0; rt < 2; ++rt)
#pragma unroll
            for (int ct = 0; ct < 2; ++ct) {
                int d = ct * 16 + nlane;
                float bk = bkv[h * HD + d];
                float bv = bkv[DIM + h * HD + d];
                int kr0 = wv * 32 + rt * 16;
#pragma unroll
                for (int i = 0; i < 8; ++i) {
                    int m = i + 8 * hs;
                    s_kh[(kr0 + m) * HD + d] = (_Float16)(accK[rt][ct][i] + bk);
                }
                v8h pv;
#pragma unroll
                for (int i = 0; i < 8; ++i) pv[i] = (_Float16)(accV[rt][ct][i] + bv);
                *(v8h*)(s_vT + (size_t)d * N2 + kr0 + 8 * hs) = pv;   // transposed
            }
    }

    // ---------------- Phase 1: Q head projection (this qblock's 128 rows) --
    // wave wv owns q rows [qb*128 + wv*16, +16); scale folded in.
    {
        v8f accQ[2] = {};
        const float* qbase = q + (size_t)b * N1 * DIM + (size_t)(qb * 128 + wv * 16) * DIM;
#pragma unroll 2
        for (int c0 = 0; c0 < DIM; c0 += 32) {
            v16h a = frag_a_f32(qbase + c0, DIM, lane);
            v16h b0 = frag_b_f32(Wq + (size_t)(h * HD) * DIM + c0, DIM, lane);
            v16h b1 = frag_b_f32(Wq + (size_t)(h * HD + 16) * DIM + c0, DIM, lane);
            accQ[0] = wmma16(a, b0, accQ[0]);
            accQ[1] = wmma16(a, b1, accQ[1]);
        }
#pragma unroll
        for (int ct = 0; ct < 2; ++ct) {
            int d = ct * 16 + nlane;
            float bqd = bq[h * HD + d];
#pragma unroll
            for (int i = 0; i < 8; ++i)
                s_qh[(wv * 16 + i + 8 * hs) * HD + d] =
                    (_Float16)((accQ[ct][i] + bqd) * SCALE);
        }
    }

    __syncthreads();   // kh / vT / qh visible to all waves

    // ---------------- Phase 2+3: logits + bias + mask + softmax ------------
    // wave wv owns q rows [wv*16, +16) (local); full N2=256 in 16 col tiles,
    // one WMMA each (K = hd = 32). Accumulators live in registers (128 VGPR).
    {
        v16h aq = frag_a_f16(s_qh + (size_t)(wv * 16) * HD, HD, lane);
        v8f acc[16];
        v8f zero = {};
#pragma unroll
        for (int j = 0; j < 16; ++j) {
            v16h bk = frag_b_f16(s_kh + (size_t)(j * 16) * HD, HD, lane);
            acc[j] = wmma16(aq, bk, zero);
        }
        // + relative-position bias (pre-gathered) + shift mask
#pragma unroll
        for (int i = 0; i < 8; ++i) {
            int qg = qb * 128 + wv * 16 + i + 8 * hs;
            const float* pb = bias_f + ((size_t)h * N1 + qg) * N2 + nlane;
            const float* pm = mask + ((size_t)wi * N1 + qg) * N2 + nlane;
#pragma unroll
            for (int j = 0; j < 16; ++j) acc[j][i] += pb[j * 16] + pm[j * 16];
        }
        // row softmax: row (i,hs) spans the 16 lanes of its half-wave group,
        // 16 elements per lane -> local reduce then shfl_xor over masks 1..8.
#pragma unroll
        for (int i = 0; i < 8; ++i) {
            float mx = acc[0][i];
#pragma unroll
            for (int j = 1; j < 16; ++j) mx = fmaxf(mx, acc[j][i]);
#pragma unroll
            for (int off = 1; off < 16; off <<= 1)
                mx = fmaxf(mx, __shfl_xor(mx, off, 32));
            float s = 0.f;
#pragma unroll
            for (int j = 0; j < 16; ++j) {
                float e = __expf(acc[j][i] - mx);
                acc[j][i] = e;
                s += e;
            }
#pragma unroll
            for (int off = 1; off < 16; off <<= 1)
                s += __shfl_xor(s, off, 32);
            float inv = 1.0f / s;
            int qr = wv * 16 + i + 8 * hs;
#pragma unroll
            for (int j = 0; j < 16; ++j)
                s_p[(size_t)qr * N2 + j * 16 + nlane] = (_Float16)(acc[j][i] * inv);
        }
    }
    // no barrier needed: each wave consumes only its own s_p rows; s_vT was
    // covered by the earlier __syncthreads().

    // ---------------- Phase 4: context = P @ V  (16 rows x 32 cols) --------
    {
        v8f ao[2] = {};
#pragma unroll 2
        for (int k0 = 0; k0 < N2; k0 += 32) {
            v16h ap = frag_a_f16(s_p + (size_t)(wv * 16) * N2 + k0, N2, lane);
            v16h b0 = frag_b_f16(s_vT + k0, N2, lane);                       // d 0..15
            v16h b1 = frag_b_f16(s_vT + (size_t)16 * N2 + k0, N2, lane);     // d 16..31
            ao[0] = wmma16(ap, b0, ao[0]);
            ao[1] = wmma16(ap, b1, ao[1]);
        }
        _Float16* xb = xh + (size_t)b * N1 * DIM;
#pragma unroll
        for (int ct = 0; ct < 2; ++ct) {
            int d = h * HD + ct * 16 + nlane;
#pragma unroll
            for (int i = 0; i < 8; ++i) {
                int qg = qb * 128 + wv * 16 + i + 8 * hs;
                xb[(size_t)qg * DIM + d] = (_Float16)ao[ct][i];
            }
        }
    }
}

// ---------------------------------------------------------------------------
// Final projection: out[r,:] = xh[r,:] @ Wp^T + bp   (65536 x 256, f32 out)
// block = 256 threads = 8 waves, 64 rows x 256 cols per block.
// wave: row tile (w&3), 8 col tiles starting at (w>>2)*8.
// ---------------------------------------------------------------------------
__global__ void proj_kernel(const _Float16* __restrict__ xh,
                            const float* __restrict__ Wp,
                            const float* __restrict__ bp,
                            float* __restrict__ out) {
    const int wv = threadIdx.x >> 5, lane = threadIdx.x & 31;
    const int nlane = lane & 15, hs = lane >> 4;
    const size_t row0 = (size_t)blockIdx.x * 64 + (size_t)(wv & 3) * 16;
    const int ct0 = (wv >> 2) * 8;

    v8f acc[8] = {};
#pragma unroll 2
    for (int c0 = 0; c0 < DIM; c0 += 32) {
        v16h a = frag_a_f16(xh + row0 * DIM + c0, DIM, lane);
#pragma unroll
        for (int t = 0; t < 8; ++t) {
            v16h bf = frag_b_f32(Wp + (size_t)((ct0 + t) * 16) * DIM + c0, DIM, lane);
            acc[t] = wmma16(a, bf, acc[t]);
        }
    }
#pragma unroll
    for (int t = 0; t < 8; ++t) {
        int col = (ct0 + t) * 16 + nlane;
        float bias = bp[col];
#pragma unroll
        for (int i = 0; i < 8; ++i) {
            size_t r = row0 + i + 8 * hs;
            out[r * DIM + col] = acc[t][i] + bias;
        }
    }
}

// ---------------------------------------------------------------------------
extern "C" void kernel_launch(void* const* d_in, const int* in_sizes, int n_in,
                              void* d_out, int out_size, void* d_ws, size_t ws_size,
                              hipStream_t stream) {
    const float* q          = (const float*)d_in[0];
    const float* kv         = (const float*)d_in[1];
    const float* mask       = (const float*)d_in[2];
    const float* Wq         = (const float*)d_in[3];
    const float* bq         = (const float*)d_in[4];
    const float* Wkv        = (const float*)d_in[5];
    const float* bkv        = (const float*)d_in[6];
    const float* bias_table = (const float*)d_in[7];
    const float* Wp         = (const float*)d_in[8];
    const float* bp         = (const float*)d_in[9];
    const int*   rel_index  = (const int*)d_in[10];

    // workspace: bias_f (8*256*256 f32 = 2 MB) | xh (256*256*256 f16 = 32 MB)
    float*    bias_f = (float*)d_ws;
    _Float16* xh     = (_Float16*)((char*)d_ws +
                        (size_t)HEADS * N1 * N2 * sizeof(float));

    bias_gather_kernel<<<(N1 * N2) / 256, 256, 0, stream>>>(bias_table, rel_index, bias_f);

    dim3 grid(2, HEADS, BATCH);
    size_t smem_bytes = (size_t)(8192 + 8192 + 4096 + 32768) * sizeof(_Float16); // 104 KB
    attn_kernel<<<grid, 256, smem_bytes, stream>>>(q, kv, mask, Wq, bq, Wkv, bkv,
                                                   bias_f, xh);

    proj_kernel<<<(BATCH * N1) / 64, 256, 0, stream>>>(xh, Wp, bp, (float*)d_out);
}
